// ReadWrapper_58652073394738
// MI455X (gfx1250) — compile-verified, matrix operand order checked by reading, not run on hardware
//
#include <hip/hip_runtime.h>
#include <hip/hip_bf16.h>

// ---------------------------------------------------------------------------
// Problem constants (from reference)
// ---------------------------------------------------------------------------
#define BSZ    4
#define CKD    64
#define HWD    1024      // 32*32
#define NMEM   16384     // T*HW
#define CVD    256
#define TOPK   30
#define KC     128       // packed K dim: [mk ; mk^2] vs [2*qk*qe ; -qe]
#define CHUNK  128       // N rows scored per workgroup iteration (8 waves x 16)
#define NSPLIT 2         // split memory axis across WGs; merged in K3
#define NSEG   (NMEM / NSPLIT)

typedef _Float16 half_t;
typedef __attribute__((ext_vector_type(16))) _Float16 v16h;
typedef __attribute__((ext_vector_type(8)))  _Float16 v8h;
typedef __attribute__((ext_vector_type(8)))  float    v8f;

// ---------------------------------------------------------------------------
// WMMA operand loader for V_WMMA_F32_16X16X32_F16 (ISA 7.12.2 layouts).
// Source is row-major [row][KC] f16. For A: row=M; for B (stored as B^T,
// [pixel][c']): row=N. Per-lane slots:
//   lane l: r = l&15, hi = l>>4
//   halfs[0..7]  = src[r][k0 + hi*8 + 0..7]
//   halfs[8..15] = src[r][k0 + 16 + hi*8 + 0..7]
// ---------------------------------------------------------------------------
__device__ __forceinline__ v16h load_wmma_ab(const half_t* __restrict__ base,
                                             int row_stride, int k0) {
  int lane = threadIdx.x & 31;
  int r    = lane & 15;
  int hi   = lane >> 4;
  const half_t* p = base + (size_t)r * row_stride + k0 + hi * 8;
  v8h lo = *(const v8h*)(p);
  v8h up = *(const v8h*)(p + 16);
  v16h out;
#pragma unroll
  for (int i = 0; i < 8; ++i) { out[i] = lo[i]; out[8 + i] = up[i]; }
  return out;
}

// ---------------------------------------------------------------------------
// K1a: Apack[b][n][c'] (f16): c'<64 -> mem_key[b][c'][n]; c'>=64 -> mk^2.
// ---------------------------------------------------------------------------
__global__ __launch_bounds__(256)
void build_apack(const float* __restrict__ mem_key, half_t* __restrict__ Apack) {
  int b  = blockIdx.y;
  int n0 = blockIdx.x * 64;
  int tid = threadIdx.x;
  __shared__ float tile[64][65];
#pragma unroll
  for (int i = 0; i < 16; ++i) {
    int c = (tid >> 6) + i * 4;
    int n = tid & 63;
    tile[c][n] = mem_key[((size_t)(b * CKD + c)) * NMEM + n0 + n];
  }
  __syncthreads();
#pragma unroll
  for (int i = 0; i < 32; ++i) {
    int j  = tid + i * 256;
    int n  = j >> 7;
    int cp = j & 127;
    float v = tile[cp & 63][n];
    if (cp >= 64) v = v * v;
    Apack[((size_t)(b * NMEM + n0 + n)) * KC + cp] = (half_t)v;
  }
}

// ---------------------------------------------------------------------------
// K1b: Bpack[b][h][c'] (f16): c'<64 -> 2*qk*qe; c'>=64 -> -qe.
//      bsq[b][h] = sum_c qe*qk*qk (fp32).
// ---------------------------------------------------------------------------
__global__ __launch_bounds__(256)
void build_bpack(const float* __restrict__ qk_g, const float* __restrict__ qe_g,
                 half_t* __restrict__ Bpack, float* __restrict__ bsq) {
  int b  = blockIdx.y;
  int h0 = blockIdx.x * 64;
  int tid = threadIdx.x;
  __shared__ float tk[64][65];
  __shared__ float te[64][65];
#pragma unroll
  for (int i = 0; i < 16; ++i) {
    int c = (tid >> 6) + i * 4;
    int h = tid & 63;
    tk[c][h] = qk_g[((size_t)(b * CKD + c)) * HWD + h0 + h];
    te[c][h] = qe_g[((size_t)(b * CKD + c)) * HWD + h0 + h];
  }
  __syncthreads();
#pragma unroll
  for (int i = 0; i < 32; ++i) {
    int j  = tid + i * 256;
    int h  = j >> 7;
    int cp = j & 127;
    int c  = cp & 63;
    float k = tk[c][h], e = te[c][h];
    float v = (cp < 64) ? (2.0f * k * e) : (-e);
    Bpack[((size_t)(b * HWD + h0 + h)) * KC + cp] = (half_t)v;
  }
  if (tid < 64) {
    int h = tid;
    float s = 0.0f;
#pragma unroll
    for (int c = 0; c < 64; ++c) s += te[c][h] * tk[c][h] * tk[c][h];
    bsq[b * HWD + h0 + h] = s;
  }
}

// ---------------------------------------------------------------------------
// K1c: mmvT[b][n][c] = mem_msk_value[b][c][n] (coalesced top-k gather rows).
// ---------------------------------------------------------------------------
__global__ __launch_bounds__(256)
void transpose_mmv(const float* __restrict__ mmv, float* __restrict__ mmvT) {
  int b  = blockIdx.z;
  int c0 = blockIdx.y * 64;
  int n0 = blockIdx.x * 64;
  int tid = threadIdx.x;
  __shared__ float tile[64][65];
#pragma unroll
  for (int i = 0; i < 16; ++i) {
    int c = (tid >> 6) + i * 4;
    int n = tid & 63;
    tile[c][n] = mmv[((size_t)(b * CVD + c0 + c)) * NMEM + n0 + n];
  }
  __syncthreads();
#pragma unroll
  for (int i = 0; i < 16; ++i) {
    int j = tid + i * 256;
    int n = j >> 6;
    int c = j & 63;
    mmvT[((size_t)(b * NMEM + n0 + n)) * CVD + c0 + c] = tile[c][n];
  }
}

// ---------------------------------------------------------------------------
// K2: fused WMMA GEMM + scaling + streaming top-30 over one N-segment.
//  - B operands register-resident for the whole kernel.
//  - A tiles double-buffered: chunk i+1 prefetched before chunk i's WMMAs,
//    so the s_wait_loadcnt lands after scan/insert/barriers, not before WMMA.
//  - 2 barriers/chunk; the serial insert phase (16 threads) overlaps the
//    next chunk's loads+WMMA (it only touches candv/topv/curmin, which the
//    next scan reads only after the scores-ready barrier).
// Writes raw top-30 values+indices per (pixel, segment); K3 merges+softmaxes.
// ---------------------------------------------------------------------------
__global__ __launch_bounds__(256)
void sim_topk_kernel(const half_t* __restrict__ Apack,
                     const half_t* __restrict__ Bpack,
                     const float*  __restrict__ bsq,
                     const float*  __restrict__ shrinkage,
                     const float*  __restrict__ valid,
                     float* __restrict__ topV, int* __restrict__ topI_g) {
  int b     = blockIdx.z;
  int split = blockIdx.y;
  int h0    = blockIdx.x * 16;
  int tid   = threadIdx.x;
  int wave  = tid >> 5;
  int lane  = tid & 31;
  int nbeg  = split * NSEG;

  __shared__ float scores[CHUNK][17];
  __shared__ float topv[16][32];
  __shared__ int   topi[16][32];
  __shared__ float curmin[16];
  __shared__ int   candcnt[16];
  __shared__ float candv[16][128];
  __shared__ int   candi[16][128];

  for (int j = tid; j < 16 * 32; j += 256) {
    topv[j >> 5][j & 31] = -3.0e38f;
    topi[j >> 5][j & 31] = 0;
  }
  if (tid < 16) { curmin[tid] = -3.0e38f; candcnt[tid] = 0; }

  const half_t* Bbase = Bpack + ((size_t)b * HWD + h0) * KC;
  v16h bop0 = load_wmma_ab(Bbase, KC, 0);
  v16h bop1 = load_wmma_ab(Bbase, KC, 32);
  v16h bop2 = load_wmma_ab(Bbase, KC, 64);
  v16h bop3 = load_wmma_ab(Bbase, KC, 96);
  float my_bsq = bsq[b * HWD + h0 + (lane & 15)];

  const float* shr  = shrinkage + (size_t)b * NMEM;
  const float* vld  = valid + (size_t)b * NMEM;
  const half_t* Arow = Apack + ((size_t)b * NMEM + nbeg + wave * 16) * KC;

  // Preload chunk 0 A operands.
  v16h a0 = load_wmma_ab(Arow, KC, 0);
  v16h a1 = load_wmma_ab(Arow, KC, 32);
  v16h a2 = load_wmma_ab(Arow, KC, 64);
  v16h a3 = load_wmma_ab(Arow, KC, 96);

  __syncthreads();  // top-k init visible

  const int NITER = NSEG / CHUNK;
  for (int it = 0; it < NITER; ++it) {
    int n0 = nbeg + it * CHUNK;

    // Prefetch chunk it+1 (last iteration redundantly reloads; discarded).
    const half_t* Anext =
        Arow + (size_t)((it + 1 < NITER) ? (it + 1) : it) * CHUNK * KC;
    v16h na0 = load_wmma_ab(Anext, KC, 0);
    v16h na1 = load_wmma_ab(Anext, KC, 32);
    v16h na2 = load_wmma_ab(Anext, KC, 64);
    v16h na3 = load_wmma_ab(Anext, KC, 96);

    v8f acc = {};
    acc = __builtin_amdgcn_wmma_f32_16x16x32_f16(false, a0, false, bop0,
                                                 (short)0, acc, false, false);
    acc = __builtin_amdgcn_wmma_f32_16x16x32_f16(false, a1, false, bop1,
                                                 (short)0, acc, false, false);
    acc = __builtin_amdgcn_wmma_f32_16x16x32_f16(false, a2, false, bop2,
                                                 (short)0, acc, false, false);
    acc = __builtin_amdgcn_wmma_f32_16x16x32_f16(false, a3, false, bop3,
                                                 (short)0, acc, false, false);

    // D layout: VGPR r -> M = r + (lane>=16 ? 8 : 0), N = lane&15.
    int mbase = (lane >> 4) * 8;
    int col   = lane & 15;
#pragma unroll
    for (int r = 0; r < 8; ++r) {
      int m = mbase + r;
      int n = n0 + wave * 16 + m;
      float s   = shr[n] * 0.125f;                 // shrinkage / sqrt(64)
      float pen = (1.0f - vld[n]) * (-60000.0f);   // validity penalty
      scores[wave * 16 + m][col] = (acc[r] - my_bsq) * s + pen;
    }
    __syncthreads();  // scores ready; prev insert also complete

    // Filter scores above current 30th-best into candidate buffer.
    {
      int p    = tid >> 4;
      int slot = tid & 15;
      float cm = curmin[p];
#pragma unroll
      for (int j = 0; j < 8; ++j) {
        int row = slot + 16 * j;
        float v = scores[row][p];
        if (v > cm) {
          int pos = atomicAdd(&candcnt[p], 1);
          candv[p][pos] = v;
          candi[p][pos] = n0 + row;
        }
      }
    }
    __syncthreads();  // scan done: scores free for next chunk, candbuf stable

    // One thread per pixel: replace-argmin insertion into top-30.
    // Overlaps next iteration's prefetch + WMMA in the other threads.
    if ((tid & 15) == 0) {
      int p = tid >> 4;
      int cnt = candcnt[p];
      float cm2 = curmin[p];
      for (int cix = 0; cix < cnt; ++cix) {
        float v = candv[p][cix];
        if (v > cm2) {
          int am = 0; float mv = topv[p][0];
          for (int q = 1; q < TOPK; ++q) {
            float tv = topv[p][q];
            if (tv < mv) { mv = tv; am = q; }
          }
          topv[p][am] = v;
          topi[p][am] = candi[p][cix];
          mv = topv[p][0];
          for (int q = 1; q < TOPK; ++q) mv = fminf(mv, topv[p][q]);
          cm2 = mv;
        }
      }
      curmin[p]  = cm2;
      candcnt[p] = 0;
    }

    // Rotate double buffer (wait for prefetch happens here, post-overlap).
    a0 = na0; a1 = na1; a2 = na2; a3 = na3;
  }
  __syncthreads();

  // Emit raw top-30 (+2 pad slots) for this segment; softmax happens in K3.
  if (tid < 16) {
    int p = tid;
    size_t base = (((size_t)b * HWD + h0 + p) * NSPLIT + split) * 32;
    for (int q = 0; q < TOPK; ++q) {
      topV[base + q]   = topv[p][q];
      topI_g[base + q] = topi[p][q];
    }
    topV[base + 30] = -3.0e38f; topV[base + 31] = -3.0e38f;
    topI_g[base + 30] = 0;      topI_g[base + 31] = 0;
  }
}

// ---------------------------------------------------------------------------
// K3: merge per-segment top-30 lists (rank selection, fully parallel),
// softmax, sparse gather of visual readout, uncertainty gate, blend.
// ---------------------------------------------------------------------------
__global__ __launch_bounds__(256)
void gather_blend(const float* __restrict__ topV, const int* __restrict__ topI_g,
                  const float* __restrict__ mmvT,
                  const float* __restrict__ pix_feat,
                  const float* __restrict__ last_mask,
                  const float* __restrict__ last_pix_feat,
                  const float* __restrict__ last_msk_value,
                  const float* __restrict__ W_uncert,
                  const float* __restrict__ b_uncert,
                  float* __restrict__ out) {
  int b   = blockIdx.y;
  int h0  = blockIdx.x * 16;
  int tid = threadIdx.x;

  const int NC = NSPLIT * 32;  // 64 merged candidates per pixel

  __shared__ float mv[16][NSPLIT * 32];
  __shared__ int   mi[16][NSPLIT * 32];
  __shared__ float eV[16][32];
  __shared__ int   eI[16][32];
  __shared__ float pmax[16];
  __shared__ float psum[16];
  __shared__ float pinv[16];
  __shared__ float vis[16][CVD];
  __shared__ float red[16][8];
  __shared__ float up[16];

  // Load both segments' candidates (pads carry -3e38 and rank last).
  for (int j = tid; j < 16 * NC; j += 256) {
    int p = j / NC, q = j % NC;
    size_t base = (((size_t)b * HWD + h0 + p) * NSPLIT) * 32 + q;
    mv[p][q] = topV[base];
    mi[p][q] = topI_g[base];
  }
  __syncthreads();

  if (tid < 16) {
    float mx = mv[tid][0];
    for (int q = 1; q < NC; ++q) mx = fmaxf(mx, mv[tid][q]);
    pmax[tid] = mx;
    psum[tid] = 0.0f;
  }
  __syncthreads();

  // Rank selection: candidate (p,q) is in top-30 iff fewer than 30 candidates
  // beat it (ties broken by slot index -> total order, no collisions).
  for (int j = tid; j < 16 * NC; j += 256) {
    int p = j / NC, q = j % NC;
    float v = mv[p][q];
    int rank = 0;
    for (int k = 0; k < NC; ++k) {
      float w = mv[p][k];
      rank += (w > v) || (w == v && k < q);
    }
    if (rank < TOPK) {
      float e = __expf(v - pmax[p]);
      eV[p][rank] = e;
      eI[p][rank] = mi[p][q];
      atomicAdd(&psum[p], e);
    }
  }
  __syncthreads();
  if (tid < 16) pinv[tid] = 1.0f / psum[tid];
  __syncthreads();

  // Sparse gather: rows of mmvT are contiguous 1KB -> coalesced across c.
  int c = tid;
#pragma unroll 1
  for (int p = 0; p < 16; ++p) {
    float v = 0.0f;
#pragma unroll
    for (int q = 0; q < TOPK; ++q)
      v += eV[p][q] * mmvT[((size_t)b * NMEM + eI[p][q]) * CVD + c];
    vis[p][c] = v * pinv[p];
  }
  __syncthreads();

  // Uncertainty logit over [last_pix_feat(256), pix_feat(256), last_mask(1), diff(256)].
  float w0 = W_uncert[c];
  float w1 = W_uncert[CVD + c];
  float w2 = W_uncert[2 * CVD + 1 + c];
#pragma unroll 1
  for (int p = 0; p < 16; ++p) {
    int h = h0 + p;
    float lpf = last_pix_feat[((size_t)b * CVD + c) * HWD + h];
    float pf  = pix_feat[((size_t)b * CVD + c) * HWD + h];
    float lmv = last_msk_value[((size_t)b * CVD + c) * HWD + h];
    float partial = w0 * lpf + w1 * pf + w2 * (vis[p][c] - lmv);
#pragma unroll
    for (int m = 16; m > 0; m >>= 1) partial += __shfl_xor(partial, m, 32);
    if ((tid & 31) == 0) red[p][tid >> 5] = partial;
  }
  __syncthreads();
  if (tid < 16) {
    int p = tid;
    float s = 0.0f;
#pragma unroll
    for (int wv = 0; wv < 8; ++wv) s += red[p][wv];
    float lm = last_mask[(size_t)b * HWD + h0 + p];
    s += lm * W_uncert[2 * CVD] + b_uncert[0];
    up[p] = 1.0f / (1.0f + __expf(-s));
  }
  __syncthreads();

  for (int j = tid; j < 16 * CVD; j += 256) {
    int cc = j >> 4;
    int p  = j & 15;
    int h  = h0 + p;
    float lmv = last_msk_value[((size_t)b * CVD + cc) * HWD + h];
    float u = up[p];
    out[((size_t)b * CVD + cc) * HWD + h] = vis[p][cc] * u + lmv * (1.0f - u);
  }
}

// ---------------------------------------------------------------------------
// Host entry. Workspace layout (bytes):
//   Apack f16 [4][16384][128]     : 16 MB  @ 0
//   Bpack f16 [4][1024][128]      :  1 MB  @ 16777216
//   bsq   f32 [4][1024]           : 16 KB  @ 17825792
//   topV  f32 [4][1024][2][32]    :  1 MB  @ 17842176
//   topI  i32 [4][1024][2][32]    :  1 MB  @ 18890752
//   mmvT  f32 [4][16384][256]     : 64 MB  @ 19939328
// ---------------------------------------------------------------------------
extern "C" void kernel_launch(void* const* d_in, const int* in_sizes, int n_in,
                              void* d_out, int out_size, void* d_ws, size_t ws_size,
                              hipStream_t stream) {
  const float* query_key       = (const float*)d_in[0];
  const float* query_selection = (const float*)d_in[1];
  const float* pix_feat        = (const float*)d_in[2];
  // d_in[3] = sensory (unused by reference)
  const float* last_mask       = (const float*)d_in[4];
  const float* last_pix_feat   = (const float*)d_in[5];
  const float* last_msk_value  = (const float*)d_in[6];
  const float* mem_key         = (const float*)d_in[7];
  const float* mem_shrinkage   = (const float*)d_in[8];
  const float* mem_msk_value   = (const float*)d_in[9];
  const float* mem_valid       = (const float*)d_in[10];
  // d_in[11] = obj_memory (unused by reference)
  const float* W_uncert        = (const float*)d_in[12];
  const float* b_uncert        = (const float*)d_in[13];

  char* ws = (char*)d_ws;
  half_t* Apack = (half_t*)(ws + 0);
  half_t* Bpack = (half_t*)(ws + 16777216);
  float*  bsq   = (float*)(ws + 17825792);
  float*  topV  = (float*)(ws + 17842176);
  int*    topI  = (int*)(ws + 18890752);
  float*  mmvT  = (float*)(ws + 19939328);

  build_apack<<<dim3(NMEM / 64, BSZ), 256, 0, stream>>>(mem_key, Apack);
  build_bpack<<<dim3(HWD / 64, BSZ), 256, 0, stream>>>(query_key, query_selection,
                                                       Bpack, bsq);
  transpose_mmv<<<dim3(NMEM / 64, CVD / 64, BSZ), 256, 0, stream>>>(mem_msk_value,
                                                                    mmvT);
  sim_topk_kernel<<<dim3(HWD / 16, NSPLIT, BSZ), 256, 0, stream>>>(
      Apack, Bpack, bsq, mem_shrinkage, mem_valid, topV, topI);
  gather_blend<<<dim3(HWD / 16, BSZ), 256, 0, stream>>>(
      topV, topI, mmvT, pix_feat, last_mask, last_pix_feat, last_msk_value,
      W_uncert, b_uncert, (float*)d_out);
}
